// RSSM_30571577213061
// MI455X (gfx1250) — matrix operand order repaired and checked by reading, not compiled, not run on hardware
//
#include <hip/hip_runtime.h>

// ---------------------------------------------------------------------------
// RSSM scan for MI455X (gfx1250): bf16 WMMA GEMMs + fp32 state carry.
// Dims: T=50 B=512 E=1024 A=64 D=1024 N=1024 S=256, SA=S+A=320
// GEMM: software-pipelined (2-stage) K loop, 32x64 strip per wave
// (2 M-tiles x 4 N-tiles = 8 WMMAs per K-step, 1.5 b128 loads per WMMA).
// ---------------------------------------------------------------------------

typedef __attribute__((ext_vector_type(16))) __bf16 v16bf;
typedef __attribute__((ext_vector_type(8)))  __bf16 bf16x8;
typedef __attribute__((ext_vector_type(8)))  float  v8f;

union FragB16 { v16bf v; bf16x8 h[2]; };

static __device__ __forceinline__ __bf16 f2bf(float f) {
  union { float f; unsigned u; } v; v.f = f;
  unsigned r = v.u + 0x7FFFu + ((v.u >> 16) & 1u);   // round-to-nearest-even
  union { unsigned short s; __bf16 b; } o; o.s = (unsigned short)(r >> 16);
  return o.b;
}

static __device__ __forceinline__ float sigmoid_f(float x) {
  return 1.0f / (1.0f + __expf(-x));
}

static __device__ __forceinline__ float softplus_f(float x) {
  return (x > 20.0f) ? x : log1pf(__expf(x));
}

// --------------------------- prep kernels ----------------------------------

// Wt[n*K + k] = bf16(W[n*K + k])   (W is [N,K] row-major; Wt is [N,K])
__global__ void wconv_transpose(const float* __restrict__ W,
                                __bf16* __restrict__ Wt, int N, int K) {
  int idx = blockIdx.x * blockDim.x + threadIdx.x;
  if (idx >= N * K) return;
  int n = idx / K, k = idx - n * K;
  Wt[(size_t)n * K + k] = f2bf(W[idx]);
}

__global__ void init_state(const float* __restrict__ s0,
                           const float* __restrict__ h0,
                           float* __restrict__ stoch,
                           float* __restrict__ h,
                           __bf16* __restrict__ hb) {
  int idx = blockIdx.x * blockDim.x + threadIdx.x;
  if (idx < 512 * 256) stoch[idx] = s0[idx];
  if (idx < 512 * 1024) {
    float v = h0[idx];
    h[idx] = v;
    hb[idx] = f2bf(v);
  }
}

// x_sa[b, 0:256] = stoch[b], x_sa[b, 256:320] = action_t[b]   (bf16)
__global__ void build_x_sa(const float* __restrict__ stoch,
                           const float* __restrict__ act_t,
                           __bf16* __restrict__ x) {
  int idx = blockIdx.x * blockDim.x + threadIdx.x;
  if (idx >= 512 * 320) return;
  int b = idx / 320, c = idx - b * 320;
  float v = (c < 256) ? stoch[b * 256 + c] : act_t[b * 64 + (c - 256)];
  x[idx] = f2bf(v);
}

// --------------------------- WMMA GEMM -------------------------------------
// out = epi(A @ Wt^T + bias).  A:[512,K] bf16 (lda), Wt:[N,K] bf16 (ld=K), ldc=N.
// Wave computes a 32(M) x 64(N) strip: 2 M-tiles x 4 N-tiles.
// WG = 8 waves -> 64(M) x 256(N).  grid = (N/256, 512/64, nz).
// blockIdx.z selects the {A,Wt,bias,out,K,lda} set so independent GEMMs
// share one launch (shorter sequential chain).
// EPI 0: f32 out. EPI 1: ELU -> bf16 out.
// Requires K/32 even: true for all GEMMs here (320, 1024, 2048).

#define LOAD_STAGE(A0F, A1F, BF)                           \
  do {                                                     \
    (A0F).h[0] = *(const bf16x8*)(ap0);                    \
    (A0F).h[1] = *(const bf16x8*)(ap0 + 16);               \
    (A1F).h[0] = *(const bf16x8*)(ap1);                    \
    (A1F).h[1] = *(const bf16x8*)(ap1 + 16);               \
    (BF)[0].h[0] = *(const bf16x8*)(bp);                   \
    (BF)[0].h[1] = *(const bf16x8*)(bp + 16);              \
    (BF)[1].h[0] = *(const bf16x8*)(bp + bj);              \
    (BF)[1].h[1] = *(const bf16x8*)(bp + bj + 16);         \
    (BF)[2].h[0] = *(const bf16x8*)(bp + 2 * bj);          \
    (BF)[2].h[1] = *(const bf16x8*)(bp + 2 * bj + 16);     \
    (BF)[3].h[0] = *(const bf16x8*)(bp + 3 * bj);          \
    (BF)[3].h[1] = *(const bf16x8*)(bp + 3 * bj + 16);     \
    ap0 += 32;                                             \
    ap1 += 32;                                             \
    bp += bstep;                                           \
  } while (0)

#define WMMA8(A0F, A1F, BF)                                                  \
  do {                                                                       \
    acc0[0] = __builtin_amdgcn_wmma_f32_16x16x32_bf16(                       \
        false, (A0F).v, false, (BF)[0].v, (short)0, acc0[0], false, false);  \
    acc1[0] = __builtin_amdgcn_wmma_f32_16x16x32_bf16(                       \
        false, (A1F).v, false, (BF)[0].v, (short)0, acc1[0], false, false);  \
    acc0[1] = __builtin_amdgcn_wmma_f32_16x16x32_bf16(                       \
        false, (A0F).v, false, (BF)[1].v, (short)0, acc0[1], false, false);  \
    acc1[1] = __builtin_amdgcn_wmma_f32_16x16x32_bf16(                       \
        false, (A1F).v, false, (BF)[1].v, (short)0, acc1[1], false, false);  \
    acc0[2] = __builtin_amdgcn_wmma_f32_16x16x32_bf16(                       \
        false, (A0F).v, false, (BF)[2].v, (short)0, acc0[2], false, false);  \
    acc1[2] = __builtin_amdgcn_wmma_f32_16x16x32_bf16(                       \
        false, (A1F).v, false, (BF)[2].v, (short)0, acc1[2], false, false);  \
    acc0[3] = __builtin_amdgcn_wmma_f32_16x16x32_bf16(                       \
        false, (A0F).v, false, (BF)[3].v, (short)0, acc0[3], false, false);  \
    acc1[3] = __builtin_amdgcn_wmma_f32_16x16x32_bf16(                       \
        false, (A1F).v, false, (BF)[3].v, (short)0, acc1[3], false, false);  \
  } while (0)

template <int EPI>
__global__ __launch_bounds__(256)
void gemm_bf16_wmma(const __bf16* __restrict__ A0, int lda0,
                    const __bf16* __restrict__ W0,
                    const float* __restrict__ bias0,
                    void* __restrict__ out0, int K0,
                    const __bf16* __restrict__ A1, int lda1,
                    const __bf16* __restrict__ W1,
                    const float* __restrict__ bias1,
                    void* __restrict__ out1, int K1,
                    int N) {
  const bool z = (blockIdx.z != 0);
  const __bf16* __restrict__ A  = z ? A1 : A0;
  const __bf16* __restrict__ Wt = z ? W1 : W0;
  const float* __restrict__ bias = z ? bias1 : bias0;
  void* __restrict__ outv = z ? out1 : out0;
  const int lda = z ? lda1 : lda0;
  const int K   = z ? K1 : K0;

  const int lane = threadIdx.x & 31;
  const int wave = threadIdx.x >> 5;
  const int m0 = blockIdx.y * 64 + (wave & 1) * 32;    // rows m0 .. m0+31
  const int n0 = blockIdx.x * 256 + (wave >> 1) * 64;  // cols n0 .. n0+63

  // A fragment addressing (16-bit A 16x32 layout):
  //   lane<16 : row=m0+lane,    K-chunks {0..7} and {16..23}
  //   lane>=16: row=m0+lane-16, K-chunks {8..15} and {24..31}
  const int arow  = m0 + (lane & 15);
  const int akoff = (lane < 16) ? 0 : 8;

  // Running load pointers (advanced by constant strides; no per-iter mul).
  const __bf16* ap0 = A + (size_t)arow * lda + akoff;         // M-tile 0
  const __bf16* ap1 = ap0 + (size_t)16 * lda;                 // M-tile 1
  const __bf16* bp  = Wt + (size_t)(n0 + (lane & 15)) * K + akoff;  // B frag: lane L = col n0+L%16, k as A
  const size_t bj = (size_t)16 * K;
  const size_t bstep = 32;

  v8f acc0[4] = {};
  v8f acc1[4] = {};
  FragB16 a0A, a1A, a0B, a1B;
  FragB16 bA[4], bB[4];

  // Software pipeline, 2 stages. S = K/32 (even for all GEMMs used here).
  const int S = K >> 5;
  LOAD_STAGE(a0A, a1A, bA);              // step 0
  for (int s = 0; s < S - 2; s += 2) {
    LOAD_STAGE(a0B, a1B, bB);            // prefetch step s+1
    WMMA8(a0A, a1A, bA);                 // compute step s
    LOAD_STAGE(a0A, a1A, bA);            // prefetch step s+2
    WMMA8(a0B, a1B, bB);                 // compute step s+1
  }
  LOAD_STAGE(a0B, a1B, bB);              // step S-1
  WMMA8(a0A, a1A, bA);                   // step S-2
  WMMA8(a0B, a1B, bB);                   // step S-1

  // C/D layout: VGPR i, lanes 0-15 -> (M=mt+i, N), lanes 16-31 -> M=mt+8+i
  const int ncol = lane & 15;
  const int rsub = (lane < 16) ? 0 : 8;
#pragma unroll
  for (int j = 0; j < 4; ++j) {
    const int n = n0 + j * 16 + ncol;
    const float bv = bias[n];
#pragma unroll
    for (int i = 0; i < 8; ++i) {
      const int ma = m0 + rsub + i;        // M-tile 0
      const int mb = ma + 16;              // M-tile 1
      float ra = acc0[j][i] + bv;
      float rb = acc1[j][i] + bv;
      if (EPI == 1) {
        ra = (ra > 0.0f) ? ra : (__expf(ra) - 1.0f);   // ELU
        rb = (rb > 0.0f) ? rb : (__expf(rb) - 1.0f);
        ((__bf16*)outv)[(size_t)ma * N + n] = f2bf(ra);
        ((__bf16*)outv)[(size_t)mb * N + n] = f2bf(rb);
      } else {
        ((float*)outv)[(size_t)ma * N + n] = ra;
        ((float*)outv)[(size_t)mb * N + n] = rb;
      }
    }
  }
}

// --------------------------- GRU gates -------------------------------------
// PyTorch GRUCell: r=sig(i_r+h_r) z=sig(i_z+h_z) n=tanh(i_n + r*h_n)
// h' = (1-z)*n + z*h.  Also: write deter to out[t], build x_post=[h'|obs].
__global__ void gru_update(const float* __restrict__ gi,
                           const float* __restrict__ gh,
                           float* __restrict__ h,
                           __bf16* __restrict__ hb,
                           __bf16* __restrict__ xpost,
                           const float* __restrict__ obs_t,
                           float* __restrict__ out_t) {
  int idx = blockIdx.x * blockDim.x + threadIdx.x;
  if (idx >= 512 * 1024) return;
  int b = idx >> 10, d = idx & 1023;
  size_t g = (size_t)b * 3072 + d;
  float r = sigmoid_f(gi[g] + gh[g]);
  float z = sigmoid_f(gi[g + 1024] + gh[g + 1024]);
  float n = tanhf(gi[g + 2048] + r * gh[g + 2048]);
  float hn = (1.0f - z) * n + z * h[idx];
  h[idx] = hn;
  __bf16 hnb = f2bf(hn);
  hb[idx] = hnb;
  xpost[(size_t)b * 2048 + d] = hnb;
  xpost[(size_t)b * 2048 + 1024 + d] = f2bf(obs_t[idx]);
  out_t[(size_t)b * 2560 + 1536 + d] = hn;   // deter slice at offset 6*S
}

// --------------------------- dist head epilogues (p & q fused) -------------
// raw:[512,512] -> mean=raw[:, :256], std=softplus(raw[:,256:])+0.1,
// stoch=mean+std*noise.  z=0: prior (base 0); z=1: posterior (base 768,
// stoch carried into next step).
__global__ void dist_head2(const float* __restrict__ praw,
                           const float* __restrict__ qraw,
                           const float* __restrict__ np_t,
                           const float* __restrict__ nq_t,
                           float* __restrict__ out_t,
                           float* __restrict__ carry) {
  int idx = blockIdx.x * blockDim.x + threadIdx.x;
  if (idx >= 2 * 512 * 256) return;
  const int zq = (idx >= 512 * 256) ? 1 : 0;
  const int i = idx - zq * 512 * 256;
  const float* raw   = zq ? qraw : praw;
  const float* noise = zq ? nq_t : np_t;
  const int b = i >> 8, s = i & 255;
  float mean = raw[(size_t)b * 512 + s];
  float stdv = softplus_f(raw[(size_t)b * 512 + 256 + s]) + 0.1f;
  float st = mean + stdv * noise[i];
  float* o = out_t + (size_t)b * 2560 + (zq ? 768 : 0);
  o[s] = mean;
  o[256 + s] = stdv;
  o[512 + s] = st;
  if (zq) carry[i] = st;
}

// --------------------------- host launch -----------------------------------

extern "C" void kernel_launch(void* const* d_in, const int* in_sizes, int n_in,
                              void* d_out, int out_size, void* d_ws, size_t ws_size,
                              hipStream_t stream) {
  (void)in_sizes; (void)n_in; (void)out_size; (void)ws_size;
  const int T = 50, B = 512, E = 1024, ACT = 64, S = 256;

  const float* obs = (const float*)d_in[0];
  const float* act = (const float*)d_in[1];
  const float* s0  = (const float*)d_in[2];
  const float* h0  = (const float*)d_in[3];
  const float* np  = (const float*)d_in[4];
  const float* nq  = (const float*)d_in[5];
  const float* Wsa = (const float*)d_in[6];
  const float* bsa = (const float*)d_in[7];
  const float* Wih = (const float*)d_in[8];
  const float* bih = (const float*)d_in[9];
  const float* Whh = (const float*)d_in[10];
  const float* bhh = (const float*)d_in[11];
  const float* Wp1 = (const float*)d_in[12];
  const float* bp1 = (const float*)d_in[13];
  const float* Wp2 = (const float*)d_in[14];
  const float* bp2 = (const float*)d_in[15];
  const float* Wq1 = (const float*)d_in[16];
  const float* bq1 = (const float*)d_in[17];
  const float* Wq2 = (const float*)d_in[18];
  const float* bq2 = (const float*)d_in[19];

  // ---- workspace carve ----
  char* wp = (char*)d_ws;
  auto carve = [&](size_t bytes) -> void* {
    void* p = wp;
    wp += (bytes + 255) & ~(size_t)255;
    return p;
  };
  __bf16* WsaT = (__bf16*)carve((size_t)320  * 1024 * 2);
  __bf16* WihT = (__bf16*)carve((size_t)1024 * 3072 * 2);
  __bf16* WhhT = (__bf16*)carve((size_t)1024 * 3072 * 2);
  __bf16* Wp1T = (__bf16*)carve((size_t)1024 * 1024 * 2);
  __bf16* Wp2T = (__bf16*)carve((size_t)1024 * 512  * 2);
  __bf16* Wq1T = (__bf16*)carve((size_t)2048 * 1024 * 2);
  __bf16* Wq2T = (__bf16*)carve((size_t)1024 * 512  * 2);
  float*  hF    = (float*) carve((size_t)512 * 1024 * 4);
  __bf16* hB    = (__bf16*)carve((size_t)512 * 1024 * 2);
  float*  stoch = (float*) carve((size_t)512 * 256 * 4);
  __bf16* xsa   = (__bf16*)carve((size_t)512 * 320 * 2);
  __bf16* saB   = (__bf16*)carve((size_t)512 * 1024 * 2);
  float*  gi    = (float*) carve((size_t)512 * 3072 * 4);
  float*  gh    = (float*) carve((size_t)512 * 3072 * 4);
  __bf16* xpost = (__bf16*)carve((size_t)512 * 2048 * 2);
  __bf16* hp    = (__bf16*)carve((size_t)512 * 1024 * 2);
  __bf16* hq    = (__bf16*)carve((size_t)512 * 1024 * 2);
  float*  praw  = (float*) carve((size_t)512 * 512 * 4);
  float*  qraw  = (float*) carve((size_t)512 * 512 * 4);

  // ---- one-time weight transpose/convert (deterministic every call) ----
  auto tr = [&](const float* W, __bf16* Wt, int n, int k) {
    int tot = n * k;
    wconv_transpose<<<(tot + 255) / 256, 256, 0, stream>>>(W, Wt, n, k);
  };
  tr(Wsa, WsaT, 1024, 320);
  tr(Wih, WihT, 3072, 1024);
  tr(Whh, WhhT, 3072, 1024);
  tr(Wp1, Wp1T, 1024, 1024);
  tr(Wp2, Wp2T, 512, 1024);
  tr(Wq1, Wq1T, 1024, 2048);
  tr(Wq2, Wq2T, 512, 1024);

  init_state<<<(512 * 1024 + 255) / 256, 256, 0, stream>>>(s0, h0, stoch, hF, hB);

  // single GEMM (grid.z = 1)
  auto gemm1 = [&](int epi, const __bf16* A, int lda, const __bf16* Wt,
                   const float* bias, void* out, int N, int K) {
    dim3 grid(N / 256, 512 / 64, 1);
    if (epi)
      gemm_bf16_wmma<1><<<grid, 256, 0, stream>>>(A, lda, Wt, bias, out, K,
                                                  A, lda, Wt, bias, out, K, N);
    else
      gemm_bf16_wmma<0><<<grid, 256, 0, stream>>>(A, lda, Wt, bias, out, K,
                                                  A, lda, Wt, bias, out, K, N);
  };
  // fused pair of independent GEMMs with same N (grid.z = 2)
  auto gemm2 = [&](int epi, const __bf16* A0, int lda0, const __bf16* W0,
                   const float* b0, void* o0, int K0,
                   const __bf16* A1, int lda1, const __bf16* W1,
                   const float* b1, void* o1, int K1, int N) {
    dim3 grid(N / 256, 512 / 64, 2);
    if (epi)
      gemm_bf16_wmma<1><<<grid, 256, 0, stream>>>(A0, lda0, W0, b0, o0, K0,
                                                  A1, lda1, W1, b1, o1, K1, N);
    else
      gemm_bf16_wmma<0><<<grid, 256, 0, stream>>>(A0, lda0, W0, b0, o0, K0,
                                                  A1, lda1, W1, b1, o1, K1, N);
  };

  for (int t = 0; t < T; ++t) {
    const float* act_t = act + (size_t)t * B * ACT;
    const float* obs_t = obs + (size_t)t * B * E;
    float* out_t = (float*)d_out + (size_t)t * B * 2560;

    build_x_sa<<<(512 * 320 + 255) / 256, 256, 0, stream>>>(stoch, act_t, xsa);
    // sa = elu([stoch|act] @ Wsa^T + bsa)
    gemm1(1, xsa, 320, WsaT, bsa, saB, 1024, 320);
    // gi = sa @ Wih^T + bih ; gh = h_{t-1} @ Whh^T + bhh  (one launch)
    gemm2(0, saB, 1024, WihT, bih, gi, 1024,
          hB, 1024, WhhT, bhh, gh, 1024, 3072);
    // GRU gates -> h_t (also writes deter slice of out, builds x_post)
    gru_update<<<(512 * 1024 + 255) / 256, 256, 0, stream>>>(gi, gh, hF, hB,
                                                             xpost, obs_t, out_t);
    // head layer 1: hp = elu(h @ Wp1^T), hq = elu(x_post @ Wq1^T)  (one launch)
    gemm2(1, hB, 1024, Wp1T, bp1, hp, 1024,
          xpost, 2048, Wq1T, bq1, hq, 2048, 1024);
    // head layer 2: praw / qraw  (one launch)
    gemm2(0, hp, 1024, Wp2T, bp2, praw, 1024,
          hq, 1024, Wq2T, bq2, qraw, 1024, 512);
    // dist epilogues for p & q (one launch); q_stoch carried to next step
    dist_head2<<<(2 * 512 * 256 + 255) / 256, 256, 0, stream>>>(
        praw, qraw, np + (size_t)t * B * S, nq + (size_t)t * B * S,
        out_t, stoch);
  }
}
